// LandmarkHSA_12919261626402
// MI455X (gfx1250) — compile-verified, matrix operand order checked by reading, not run on hardware
//
#include <hip/hip_runtime.h>
#include <hip/hip_bf16.h>

// ---------------- problem constants ----------------
constexpr int L = 2048, D = 2048, HD = 64;
constexpr int SWA_H = 24, SWA_KV = 6, HSA_H = 8, HSA_KV = 2;
constexpr int CHUNK = 64, TOPK = 16, SW = 512;
constexpr int NC = L / CHUNK;            // 32 chunks
constexpr float SCALE = 0.125f;          // 64^-0.5
constexpr float NEGC  = -1.0e9f;

typedef __attribute__((ext_vector_type(16))) _Float16 v16h;
typedef __attribute__((ext_vector_type(8)))  _Float16 v8h;
typedef __attribute__((ext_vector_type(4)))  _Float16 v4h;
typedef __attribute__((ext_vector_type(8)))  float    v8f;

// K index inside a 16-bit 16x32 A (or mirrored B) fragment for element i, lane-half g:
// {8g..8g+7} and {16+8g..16+8g+7}  -> two contiguous 16B runs, vectorizes to 2x b128
__device__ __forceinline__ int akof(int i, int g) {
    return ((i & 8) << 1) + (g << 3) + (i & 7);
}

__device__ __forceinline__ v8f wmma(const v16h& a, const v16h& b, const v8f& c) {
    return __builtin_amdgcn_wmma_f32_16x16x32_f16(false, a, false, b, (short)0, c, false, false);
}

// reductions across the 16 lanes of a half-wave (xor masks 1..8 stay within a half)
__device__ __forceinline__ float rowmax16(float v) {
    for (int m = 8; m >= 1; m >>= 1) v = fmaxf(v, __shfl_xor(v, m, 32));
    return v;
}
__device__ __forceinline__ float rowsum16(float v) {
    for (int m = 8; m >= 1; m >>= 1) v += __shfl_xor(v, m, 32);
    return v;
}

// ---------------- f32 -> f16 convert ----------------
__global__ void cvt_f32_f16(const float* __restrict__ in, _Float16* __restrict__ out, int n) {
    int i = blockIdx.x * 256 + threadIdx.x;
    if (i < n) out[i] = (_Float16)in[i];
}

// ---------------- tiled WMMA GEMM: C[M,N] = A16[M,K] * Bf32[K,N] ----------------
// block = 128 threads (4 waves); block tile 64x64; each wave computes 16x64.
// B is staged TRANSPOSED in LDS so both fragment reads are 2x ds_load_b128.
__global__ void gemm_f16w(const _Float16* __restrict__ A, const float* __restrict__ B,
                          float* __restrict__ C, int M, int N, int K) {
    __shared__ _Float16 As[64][40];     // [m][k]
    __shared__ _Float16 BsT[64][40];    // [n][k]
    const int t = threadIdx.x;
    const int lane = t & 31, w = t >> 5;
    const int mrow = lane & 15, g = lane >> 4;
    const int mbase = blockIdx.y * 64, nbase = blockIdx.x * 64;
    v8f zer = {0.f, 0.f, 0.f, 0.f, 0.f, 0.f, 0.f, 0.f};
    v8f acc[4];
    for (int i = 0; i < 4; i++) acc[i] = zer;

    const int arow_ld = t >> 1, ac0 = (t & 1) * 16;       // A staging coords
    const int bk0 = (t & 7) * 4, bn0 = (t >> 3) * 4;      // B staging coords (4x4 micro-transpose)

    for (int k0 = 0; k0 < K; k0 += 32) {
        {   // stage A tile (64x32 f16), vector copies
            const v8h* ap = (const v8h*)(A + (size_t)(mbase + arow_ld) * K + k0 + ac0);
            *(v8h*)&As[arow_ld][ac0]     = ap[0];
            *(v8h*)&As[arow_ld][ac0 + 8] = ap[1];
        }
        {   // stage B tile (32x64 f32) transposed -> BsT[n][k], f16
            float rr[4][4];
            for (int dk = 0; dk < 4; dk++) {
                const float4 f = *(const float4*)(B + (size_t)(k0 + bk0 + dk) * N + nbase + bn0);
                rr[dk][0] = f.x; rr[dk][1] = f.y; rr[dk][2] = f.z; rr[dk][3] = f.w;
            }
            for (int dn = 0; dn < 4; dn++) {
                v4h val;
                val[0] = (_Float16)rr[0][dn]; val[1] = (_Float16)rr[1][dn];
                val[2] = (_Float16)rr[2][dn]; val[3] = (_Float16)rr[3][dn];
                *(v4h*)&BsT[bn0 + dn][bk0] = val;
            }
        }
        if (k0 + 32 < K) {   // prefetch next K panel (global_prefetch_b8)
            __builtin_prefetch(B + (size_t)(k0 + 32 + bk0) * N + nbase + bn0, 0, 1);
            __builtin_prefetch(A + (size_t)(mbase + arow_ld) * K + k0 + 32 + ac0, 0, 1);
        }
        __syncthreads();
        v16h a;
        const int arow = w * 16 + mrow;
        for (int i = 0; i < 16; i++) a[i] = As[arow][akof(i, g)];
        for (int nt = 0; nt < 4; nt++) {
            v16h b;
            const int n = nt * 16 + mrow;
            for (int i = 0; i < 16; i++) b[i] = BsT[n][akof(i, g)];
            acc[nt] = wmma(a, b, acc[nt]);
        }
        __syncthreads();
    }
    for (int nt = 0; nt < 4; nt++)
        for (int r = 0; r < 8; r++) {
            int row = mbase + w * 16 + r + 8 * g;
            C[(size_t)row * N + nbase + nt * 16 + mrow] = acc[nt][r];
        }
}

// ---------------- per-head epilogue: rms_norm (+rope) -> f16 [H][L][64] (+optional [H][64][L]) ----
__global__ void head_post(const float* __restrict__ X, const float* __restrict__ nw,
                          const float* __restrict__ cosT, const float* __restrict__ sinT,
                          _Float16* __restrict__ Y, _Float16* __restrict__ YT,
                          int H, int do_norm, int do_rope) {
    const int l = blockIdx.x, h = blockIdx.y, d = threadIdx.x;
    __shared__ float red[64];
    __shared__ float xs[64];
    float x = X[(size_t)l * (H * HD) + h * HD + d];
    float xn = x;
    if (do_norm) {
        red[d] = x * x;
        __syncthreads();
        for (int s = 32; s > 0; s >>= 1) { if (d < s) red[d] += red[d + s]; __syncthreads(); }
        xn = x * rsqrtf(red[0] * (1.0f / HD) + 1e-6f) * nw[d];
    }
    float o = xn;
    if (do_rope) {
        xs[d] = xn;
        __syncthreads();
        float rh = (d < 32) ? -xs[d + 32] : xs[d - 32];
        o = xn * cosT[(size_t)l * HD + d] + rh * sinT[(size_t)l * HD + d];
    }
    if (Y)  Y[((size_t)h * L + l) * HD + d] = (_Float16)o;
    if (YT) YT[((size_t)h * HD + d) * L + l] = (_Float16)o;   // transposed copy (for P@V operand)
}

// ---------------- landmarks: chunk means of normed hk ----------------
__global__ void lmk_kernel(const _Float16* __restrict__ hk16, _Float16* __restrict__ lmk16) {
    const int c = blockIdx.x % NC, kv = blockIdx.x / NC, d = threadIdx.x;
    const _Float16* p = hk16 + ((size_t)kv * L + c * CHUNK) * HD + d;
    float s = 0.f;
    for (int j = 0; j < CHUNK; j++) s += (float)p[j * HD];
    lmk16[((size_t)kv * NC + c) * HD + d] = (_Float16)(s * (1.0f / CHUNK));
}

// ---------------- top-k chunk selection -> 32-bit mask per (h,q) ----------------
__global__ void topk_kernel(const _Float16* __restrict__ hq16, const _Float16* __restrict__ lmk16,
                            unsigned* __restrict__ sel) {
    const int idx = blockIdx.x * 128 + threadIdx.x;
    if (idx >= HSA_H * L) return;
    const int h = idx / L, q = idx % L, kv = h >> 2;
    const _Float16* qp = hq16 + ((size_t)h * L + q) * HD;
    float ret[NC];
    for (int c = 0; c < NC; c++) {
        const _Float16* lp = lmk16 + ((size_t)kv * NC + c) * HD;
        float s = 0.f;
        for (int d = 0; d < HD; d++) s += (float)qp[d] * (float)lp[d];
        ret[c] = s * SCALE;
    }
    int t = q - SW + 1;                                     // floor((q-511)/64)
    int co = (t >= 0) ? (t >> 6) : -((-t + 63) >> 6);
    if (co > NC) co = NC;
    bool used[NC];
    for (int c = 0; c < NC; c++) used[c] = false;
    unsigned bits = 0;
    for (int it = 0; it < TOPK; it++) {
        int best = -1; float bv = -3.0e38f;
        for (int c = 0; c < NC; c++) {
            float v = (c < co) ? ret[c] : NEGC;
            if (!used[c] && v > bv) { bv = v; best = c; }
        }
        if (best >= 0) { used[best] = true; if (best < co) bits |= (1u << best); }
    }
    sel[idx] = bits;
}

// ---------------- one 16q x 32k flash-attention tile (wave32 WMMA) ----------------
// kmat: [L][64] per kv-head;  vTmat: [64][L] per kv-head (transposed -> contiguous fragments)
__device__ __forceinline__ void attn_tile(
    const v16h& aq0, const v16h& aq1,
    const _Float16* __restrict__ kmat, const _Float16* __restrict__ vTmat,
    int kb, int qbase, int lane, _Float16 (*Pl)[40],
    int use_window, int chunk_c, const unsigned* selS,
    v8f accO[4], float mrun[8], float lrun[8]) {
    const int mrow = lane & 15, g = lane >> 4;
    v8f zer = {0.f, 0.f, 0.f, 0.f, 0.f, 0.f, 0.f, 0.f};
    v8f sc[2];
    for (int ni = 0; ni < 2; ni++) {
        int key = kb + ni * 16 + mrow;
        int keyc = (key >= L) ? (L - 1) : key;          // clamp row ptr; masked below
        const _Float16* kp = kmat + (size_t)keyc * HD;
        v16h b0, b1;
        for (int i = 0; i < 16; i++) { int kk = akof(i, g); b0[i] = kp[kk]; b1[i] = kp[32 + kk]; }
        sc[ni] = wmma(aq0, b0, zer);
        sc[ni] = wmma(aq1, b1, sc[ni]);
    }
    for (int r = 0; r < 8; r++) {
        const int row = qbase + r + 8 * g;
        const int key0 = kb + mrow, key1 = kb + 16 + mrow;
        bool v0, v1;
        if (use_window) {
            int r0 = row - key0, r1 = row - key1;
            v0 = (r0 >= 0) && (r0 < SW);
            v1 = (r1 >= 0) && (r1 < SW);
        } else {
            bool sv = (selS[r + 8 * g] >> chunk_c) & 1u;
            v0 = sv; v1 = sv;
        }
        float s0 = v0 ? sc[0][r] * SCALE : NEGC;
        float s1 = v1 ? sc[1][r] * SCALE : NEGC;
        float mnew = fmaxf(mrun[r], rowmax16(fmaxf(s0, s1)));
        float alpha = __expf(mrun[r] - mnew);
        float p0 = __expf(s0 - mnew), p1 = __expf(s1 - mnew);
        Pl[r + 8 * g][mrow]      = (_Float16)p0;
        Pl[r + 8 * g][16 + mrow] = (_Float16)p1;
        lrun[r] = lrun[r] * alpha + rowsum16(p0 + p1);
        mrun[r] = mnew;
        for (int nt = 0; nt < 4; nt++) accO[nt][r] = accO[nt][r] * alpha;
    }
    __syncthreads();                                    // single-wave: LDS visibility
    v16h pa;
    for (int i = 0; i < 16; i++) pa[i] = Pl[mrow][akof(i, g)];
    // whole-tile clamp keeps the runs contiguous; OOB keys have P == 0 exactly
    const int kbv = (kb + 32 <= L) ? kb : (L - 32);
    for (int nt = 0; nt < 4; nt++) {
        v16h bv;
        const _Float16* vp = vTmat + (size_t)(nt * 16 + mrow) * L + kbv;
        for (int i = 0; i < 16; i++) bv[i] = vp[akof(i, g)];
        accO[nt] = wmma(pa, bv, accO[nt]);
    }
    __syncthreads();
}

// ---------------- SWA: 24 heads, window 512 ----------------
__global__ void swa_attn(const _Float16* __restrict__ q16, const _Float16* __restrict__ k16,
                         const _Float16* __restrict__ vT16, float* __restrict__ out) {
    const int h = blockIdx.y, kv = h >> 2;
    const int qbase = blockIdx.x * 16;
    const int lane = threadIdx.x & 31, mrow = lane & 15, g = lane >> 4;
    __shared__ _Float16 Pl[16][40];
    const _Float16* kmat  = k16  + (size_t)kv * L * HD;
    const _Float16* vTmat = vT16 + (size_t)kv * HD * L;
    const _Float16* qp = q16 + ((size_t)h * L + qbase) * HD;
    v16h aq0, aq1;
    for (int i = 0; i < 16; i++) {
        int k = akof(i, g);
        aq0[i] = qp[(size_t)mrow * HD + k];
        aq1[i] = qp[(size_t)mrow * HD + 32 + k];
    }
    v8f zer = {0.f, 0.f, 0.f, 0.f, 0.f, 0.f, 0.f, 0.f};
    v8f accO[4]; float mrun[8], lrun[8];
    for (int i = 0; i < 4; i++) accO[i] = zer;
    for (int r = 0; r < 8; r++) { mrun[r] = -1.0e30f; lrun[r] = 0.f; }
    int kstart = qbase - SW + 1; if (kstart < 0) kstart = 0; kstart &= ~31;
    for (int kb = kstart; kb <= qbase + 15; kb += 32)
        attn_tile(aq0, aq1, kmat, vTmat, kb, qbase, lane, Pl, 1, 0, nullptr, accO, mrun, lrun);
    const int colbase = h * HD;
    for (int nt = 0; nt < 4; nt++)
        for (int r = 0; r < 8; r++) {
            int row = qbase + r + 8 * g;
            out[(size_t)row * D + colbase + nt * 16 + mrow] = accO[nt][r] / lrun[r];
        }
}

// ---------------- HSA: 8 heads, selected chunks + local window ----------------
__global__ void hsa_attn(const _Float16* __restrict__ hq16, const _Float16* __restrict__ hk16,
                         const _Float16* __restrict__ hvT16, const unsigned* __restrict__ sel,
                         float* __restrict__ out) {
    const int h = blockIdx.y, kv = h >> 2;
    const int qbase = blockIdx.x * 16;
    const int lane = threadIdx.x & 31, mrow = lane & 15, g = lane >> 4;
    __shared__ unsigned selS[16];
    __shared__ _Float16 Pl[16][40];
    if (threadIdx.x < 16) selS[threadIdx.x] = sel[(size_t)h * L + qbase + threadIdx.x];
    __syncthreads();
    unsigned uni = 0;
    for (int i = 0; i < 16; i++) uni |= selS[i];
    const _Float16* kmat  = hk16  + (size_t)kv * L * HD;
    const _Float16* vTmat = hvT16 + (size_t)kv * HD * L;
    const _Float16* qp = hq16 + ((size_t)h * L + qbase) * HD;
    v16h aq0, aq1;
    for (int i = 0; i < 16; i++) {
        int k = akof(i, g);
        aq0[i] = qp[(size_t)mrow * HD + k];
        aq1[i] = qp[(size_t)mrow * HD + 32 + k];
    }
    v8f zer = {0.f, 0.f, 0.f, 0.f, 0.f, 0.f, 0.f, 0.f};
    v8f accO[4]; float mrun[8], lrun[8];
    for (int i = 0; i < 4; i++) accO[i] = zer;
    for (int r = 0; r < 8; r++) { mrun[r] = -1.0e30f; lrun[r] = 0.f; }
    // phase 1: selected landmark chunks (always strictly before the local window)
    for (int c = 0; c < NC; c++) {
        if (!((uni >> c) & 1u)) continue;
        attn_tile(aq0, aq1, kmat, vTmat, c * CHUNK,      qbase, lane, Pl, 0, c, selS, accO, mrun, lrun);
        attn_tile(aq0, aq1, kmat, vTmat, c * CHUNK + 32, qbase, lane, Pl, 0, c, selS, accO, mrun, lrun);
    }
    // phase 2: local sliding window (guarantees every row has real scores)
    int kstart = qbase - SW + 1; if (kstart < 0) kstart = 0; kstart &= ~31;
    for (int kb = kstart; kb <= qbase + 15; kb += 32)
        attn_tile(aq0, aq1, kmat, vTmat, kb, qbase, lane, Pl, 1, 0, selS, accO, mrun, lrun);
    const int colbase = (SWA_H + h) * HD;
    for (int nt = 0; nt < 4; nt++)
        for (int r = 0; r < 8; r++) {
            int row = qbase + r + 8 * g;
            out[(size_t)row * D + colbase + nt * 16 + mrow] = accO[nt][r] / lrun[r];
        }
}

// ---------------- orchestration ----------------
extern "C" void kernel_launch(void* const* d_in, const int* in_sizes, int n_in,
                              void* d_out, int out_size, void* d_ws, size_t ws_size,
                              hipStream_t stream) {
    const float* hidden = (const float*)d_in[0];
    const float* cosT   = (const float*)d_in[1];
    const float* sinT   = (const float*)d_in[2];
    const float* w_q    = (const float*)d_in[3];
    const float* w_k    = (const float*)d_in[4];
    const float* w_v    = (const float*)d_in[5];
    const float* w_hq   = (const float*)d_in[6];
    const float* w_hk   = (const float*)d_in[7];
    const float* w_hv   = (const float*)d_in[8];
    const float* w_o    = (const float*)d_in[9];
    const float* qnw    = (const float*)d_in[10];
    const float* knw    = (const float*)d_in[11];
    float* out = (float*)d_out;

    char* ws = (char*)d_ws;
    size_t off = 0;
    auto carve = [&](size_t bytes) -> char* {
        char* p = ws + off;
        off += (bytes + 255) & ~(size_t)255;
        return p;
    };
    _Float16* h16   = (_Float16*)carve((size_t)L * D * 2);
    float*    proj32= (float*)   carve((size_t)L * (SWA_H * HD) * 4);   // max proj, reused
    _Float16* q16   = (_Float16*)carve((size_t)SWA_H * L * HD * 2);
    _Float16* k16   = (_Float16*)carve((size_t)SWA_KV * L * HD * 2);
    _Float16* vT16  = (_Float16*)carve((size_t)SWA_KV * HD * L * 2);    // transposed V
    _Float16* hq16  = (_Float16*)carve((size_t)HSA_H * L * HD * 2);
    _Float16* hk16  = (_Float16*)carve((size_t)HSA_KV * L * HD * 2);
    _Float16* hvT16 = (_Float16*)carve((size_t)HSA_KV * HD * L * 2);    // transposed HV
    _Float16* lmk16 = (_Float16*)carve((size_t)HSA_KV * NC * HD * 2);
    unsigned* sel   = (unsigned*)carve((size_t)HSA_H * L * 4);
    float*    attn32= (float*)   carve((size_t)L * D * 4);
    _Float16* ao16  = (_Float16*)proj32;   // reuse projection scratch (8MB <= 12MB)

    cvt_f32_f16<<<(L * D + 255) / 256, 256, 0, stream>>>(hidden, h16, L * D);

    auto proj = [&](const float* w, int Hh, int dn, int dr, const float* nw,
                    _Float16* dst, _Float16* dstT) {
        dim3 gg(Hh * HD / 64, L / 64);
        gemm_f16w<<<gg, 128, 0, stream>>>(h16, w, proj32, L, Hh * HD, D);
        dim3 pg(L, Hh);
        head_post<<<pg, 64, 0, stream>>>(proj32, nw, cosT, sinT, dst, dstT, Hh, dn, dr);
    };
    proj(w_q,  SWA_H,  1, 1, qnw, q16,  nullptr);
    proj(w_k,  SWA_KV, 1, 1, knw, k16,  nullptr);
    proj(w_v,  SWA_KV, 0, 0, qnw, nullptr, vT16);
    proj(w_hq, HSA_H,  1, 0, qnw, hq16, nullptr);
    proj(w_hk, HSA_KV, 1, 0, knw, hk16, nullptr);
    proj(w_hv, HSA_KV, 0, 0, qnw, nullptr, hvT16);

    lmk_kernel<<<HSA_KV * NC, 64, 0, stream>>>(hk16, lmk16);
    topk_kernel<<<(HSA_H * L) / 128, 128, 0, stream>>>(hq16, lmk16, sel);

    swa_attn<<<dim3(L / 16, SWA_H), 32, 0, stream>>>(q16, k16, vT16, attn32);
    hsa_attn<<<dim3(L / 16, HSA_H), 32, 0, stream>>>(hq16, hk16, hvT16, sel, attn32);

    cvt_f32_f16<<<(L * D + 255) / 256, 256, 0, stream>>>(attn32, ao16, L * D);
    gemm_f16w<<<dim3(D / 64, L / 64), 128, 0, stream>>>(ao16, w_o, out, L, D, D);
}
